// GCN_2LAYERS_58033598103986
// MI455X (gfx1250) — compile-verified, matrix operand order checked by reading, not run on hardware
//
#include <hip/hip_runtime.h>
#include <hip/hip_bf16.h>

typedef __attribute__((ext_vector_type(16))) _Float16 v16h;
typedef __attribute__((ext_vector_type(8)))  float    v8f;

#define FIN 128
#define HID 64

// ---------------------------------------------------------------------------
// Degree build: deg = 1 + in-degree(dst)
// ---------------------------------------------------------------------------
__global__ void k_deg_init(float* __restrict__ deg, int n) {
    int i = blockIdx.x * blockDim.x + threadIdx.x;
    if (i < n) deg[i] = 1.0f;
}

__global__ void k_deg_scatter(const int* __restrict__ dst, float* __restrict__ deg, int E) {
    int i = blockIdx.x * blockDim.x + threadIdx.x;
    if (i < E) atomicAdd(&deg[dst[i]], 1.0f);
}

__global__ void k_rsqrt_inplace(float* __restrict__ deg, int n) {
    int i = blockIdx.x * blockDim.x + threadIdx.x;
    if (i < n) deg[i] = rsqrtf(deg[i]);
}

// ---------------------------------------------------------------------------
// GEMM1 via WMMA: xw1[N,64] = f16(x[N,128]) @ f16(W1[128,64]), f32 accumulate.
// One wave -> one 16-row M-tile x full 64-col output strip.
// Fragment layouts per CDNA5 ISA 7.12.2 (16-bit A 16x32, f32 C/D 16x16).
// ---------------------------------------------------------------------------
__global__ void k_gemm1_wmma(const float* __restrict__ x, const float* __restrict__ W1,
                             float* __restrict__ xw1, int n) {
    const int lane    = threadIdx.x & 31;
    const int wave    = threadIdx.x >> 5;
    const int mtile   = blockIdx.x * (blockDim.x >> 5) + wave;
    const int half_hi = lane >> 4;          // 0: K-group low, 1: K-group high
    const int l16     = lane & 15;
    const int kb      = half_hi * 8;

    // K offsets for 16 fragment elements of this lane within a 32-wide K step
    int koff[16];
#pragma unroll
    for (int e = 0; e < 16; ++e) {
        const int vg = e >> 1, p = e & 1;
        koff[e] = (vg < 4) ? (kb + 2 * vg + p) : (16 + kb + 2 * (vg - 4) + p);
    }

    const int row  = mtile * 16 + l16;
    const int arow = (row < n) ? row : (n - 1);   // clamp tail tile (not stored)

    v8f acc[4] = {v8f{}, v8f{}, v8f{}, v8f{}};

#pragma unroll
    for (int ks = 0; ks < FIN; ks += 32) {
        // A fragment: row = arow, K = ks + koff[e]
        v16h a;
        const float* ap = x + (size_t)arow * FIN + ks;
#pragma unroll
        for (int e = 0; e < 16; ++e) a[e] = (_Float16)ap[koff[e]];

#pragma unroll
        for (int nt = 0; nt < 4; ++nt) {
            // B fragment: col = nt*16 + l16, K = ks + koff[e]  (W1 row-major [FIN,HID])
            v16h b;
            const int col = nt * 16 + l16;
#pragma unroll
            for (int e = 0; e < 16; ++e)
                b[e] = (_Float16)W1[(size_t)(ks + koff[e]) * HID + col];

            acc[nt] = __builtin_amdgcn_wmma_f32_16x16x32_f16(
                /*neg_a=*/false, a, /*neg_b=*/false, b,
                /*c_mod=*/(short)0, acc[nt], /*reuse_a=*/false, /*reuse_b=*/false);
        }
    }

    // D layout: VGPR v -> M = mtile*16 + v + half_hi*8, N = nt*16 + l16
#pragma unroll
    for (int nt = 0; nt < 4; ++nt) {
#pragma unroll
        for (int v = 0; v < 8; ++v) {
            const int orow = mtile * 16 + v + half_hi * 8;
            if (orow < n) xw1[(size_t)orow * HID + nt * 16 + l16] = acc[nt][v];
        }
    }
}

// ---------------------------------------------------------------------------
// Layer-1 aggregation
// ---------------------------------------------------------------------------
__global__ void k_self1(const float* __restrict__ xw1, const float* __restrict__ dinv,
                        float* __restrict__ agg1, int n) {
    long long i = (long long)blockIdx.x * blockDim.x + threadIdx.x;
    if (i < (long long)n * HID) {
        const int node = (int)(i >> 6);          // /HID
        const float di = dinv[node];
        agg1[i] = xw1[i] * di * di;
    }
}

__global__ void k_scatter1(const int* __restrict__ src, const int* __restrict__ dst,
                           const float* __restrict__ dinv, const float* __restrict__ xw1,
                           float* __restrict__ agg1, int E) {
    long long t = (long long)blockIdx.x * blockDim.x + threadIdx.x;
    const int e = (int)(t >> 4);
    const int g = (int)(t & 15);
    if (e >= E) return;
    const int s = src[e], d = dst[e];
    const float norm = dinv[s] * dinv[d];
    const float4 v = ((const float4*)(xw1 + (size_t)s * HID))[g];
    float* out = agg1 + (size_t)d * HID + g * 4;
    atomicAdd(out + 0, v.x * norm);
    atomicAdd(out + 1, v.y * norm);
    atomicAdd(out + 2, v.z * norm);
    atomicAdd(out + 3, v.w * norm);
}

__global__ void k_act1(float* __restrict__ agg1, const float* __restrict__ b1, int n) {
    long long i = (long long)blockIdx.x * blockDim.x + threadIdx.x;
    if (i < (long long)n * HID) {
        const int f = (int)(i & (HID - 1));
        agg1[i] = tanhf(agg1[i] + b1[f]);        // agg1 becomes h1 in place
    }
}

// ---------------------------------------------------------------------------
// Layer 2: output width 2 -> plain VALU GEMM, then aggregate
// ---------------------------------------------------------------------------
__global__ void k_gemm2(const float* __restrict__ h1, const float* __restrict__ W2,
                        float* __restrict__ xw2, int n) {
    int node = blockIdx.x * blockDim.x + threadIdx.x;
    if (node >= n) return;
    const float* hp = h1 + (size_t)node * HID;
    float o0 = 0.f, o1 = 0.f;
#pragma unroll 8
    for (int f = 0; f < HID; ++f) {
        const float hv = hp[f];
        o0 += hv * W2[f * 2 + 0];
        o1 += hv * W2[f * 2 + 1];
    }
    xw2[(size_t)node * 2 + 0] = o0;
    xw2[(size_t)node * 2 + 1] = o1;
}

__global__ void k_self2(const float* __restrict__ xw2, const float* __restrict__ dinv,
                        float* __restrict__ agg2, int n) {
    int node = blockIdx.x * blockDim.x + threadIdx.x;
    if (node >= n) return;
    const float di2 = dinv[node] * dinv[node];
    agg2[(size_t)node * 2 + 0] = xw2[(size_t)node * 2 + 0] * di2;
    agg2[(size_t)node * 2 + 1] = xw2[(size_t)node * 2 + 1] * di2;
}

__global__ void k_scatter2(const int* __restrict__ src, const int* __restrict__ dst,
                           const float* __restrict__ dinv, const float* __restrict__ xw2,
                           float* __restrict__ agg2, int E) {
    int e = blockIdx.x * blockDim.x + threadIdx.x;
    if (e >= E) return;
    const int s = src[e], d = dst[e];
    const float norm = dinv[s] * dinv[d];
    atomicAdd(&agg2[(size_t)d * 2 + 0], xw2[(size_t)s * 2 + 0] * norm);
    atomicAdd(&agg2[(size_t)d * 2 + 1], xw2[(size_t)s * 2 + 1] * norm);
}

// ---------------------------------------------------------------------------
// Fused: h = tanh(agg2 + b2); out = h @ Wc + bc
// ---------------------------------------------------------------------------
__global__ void k_final(const float* __restrict__ agg2, const float* __restrict__ b2,
                        const float* __restrict__ Wc, const float* __restrict__ bc,
                        float* __restrict__ out, float* __restrict__ hout, int n) {
    int i = blockIdx.x * blockDim.x + threadIdx.x;
    if (i >= n) return;
    const float h0 = tanhf(agg2[(size_t)i * 2 + 0] + b2[0]);
    const float h1 = tanhf(agg2[(size_t)i * 2 + 1] + b2[1]);
    hout[(size_t)i * 2 + 0] = h0;
    hout[(size_t)i * 2 + 1] = h1;
#pragma unroll
    for (int c = 0; c < 4; ++c)
        out[(size_t)i * 4 + c] = h0 * Wc[0 * 4 + c] + h1 * Wc[1 * 4 + c] + bc[c];
}

// ---------------------------------------------------------------------------
extern "C" void kernel_launch(void* const* d_in, const int* in_sizes, int n_in,
                              void* d_out, int out_size, void* d_ws, size_t ws_size,
                              hipStream_t stream) {
    const float* x   = (const float*)d_in[0];
    const int*   ei  = (const int*)d_in[1];
    const float* W1  = (const float*)d_in[2];
    const float* b1  = (const float*)d_in[3];
    const float* W2  = (const float*)d_in[4];
    const float* b2  = (const float*)d_in[5];
    const float* Wc  = (const float*)d_in[6];
    const float* bc  = (const float*)d_in[7];

    const int N = in_sizes[0] / FIN;
    const int E = in_sizes[1] / 2;
    const int* src = ei;
    const int* dst = ei + E;

    // workspace layout (floats): dinv[N] | xw1[N*64] | agg1/h1[N*64] | xw2[N*2] | agg2[N*2]
    float* ws   = (float*)d_ws;
    float* dinv = ws;
    float* xw1  = dinv + N;
    float* agg1 = xw1 + (size_t)N * HID;
    float* xw2  = agg1 + (size_t)N * HID;
    float* agg2 = xw2 + (size_t)N * 2;

    float* out  = (float*)d_out;          // [N,4]
    float* hout = out + (size_t)N * 4;    // [N,2]

    const int B = 256;
    const int gN   = (N + B - 1) / B;
    const int gE   = (E + B - 1) / B;
    const long long nh = (long long)N * HID;
    const int gNH  = (int)((nh + B - 1) / B);
    const long long e16 = (long long)E * 16;
    const int gE16 = (int)((e16 + B - 1) / B);
    const int mtiles = (N + 15) / 16;
    const int gGemm  = (mtiles + 7) / 8;  // 8 waves (mtiles) per 256-thread block

    // degree / norm
    k_deg_init<<<gN, B, 0, stream>>>(dinv, N);
    k_deg_scatter<<<gE, B, 0, stream>>>(dst, dinv, E);
    k_rsqrt_inplace<<<gN, B, 0, stream>>>(dinv, N);

    // layer 1
    k_gemm1_wmma<<<gGemm, B, 0, stream>>>(x, W1, xw1, N);
    k_self1<<<gNH, B, 0, stream>>>(xw1, dinv, agg1, N);
    k_scatter1<<<gE16, B, 0, stream>>>(src, dst, dinv, xw1, agg1, E);
    k_act1<<<gNH, B, 0, stream>>>(agg1, b1, N);   // agg1 -> h1

    // layer 2
    k_gemm2<<<gN, B, 0, stream>>>(agg1, W2, xw2, N);
    k_self2<<<gN, B, 0, stream>>>(xw2, dinv, agg2, N);
    k_scatter2<<<gE, B, 0, stream>>>(src, dst, dinv, xw2, agg2, E);

    // classifier + outputs
    k_final<<<gN, B, 0, stream>>>(agg2, b2, Wc, bc, out, hout, N);
}
